// FineTransformer_16947940950622
// MI455X (gfx1250) — matrix operand order repaired and checked by reading, not compiled, not run on hardware
//
#include <hip/hip_runtime.h>
#include <cmath>

// ---------------------------------------------------------------------------
// Types / WMMA helpers (CDNA5 gfx1250, wave32)
// ---------------------------------------------------------------------------
typedef _Float16 v16h __attribute__((ext_vector_type(16)));
typedef _Float16 v8h  __attribute__((ext_vector_type(8)));
typedef _Float16 v2h  __attribute__((ext_vector_type(2)));
typedef float    v8f  __attribute__((ext_vector_type(8)));

#define WMMA_F16(A, B, C) \
  __builtin_amdgcn_wmma_f32_16x16x32_f16(false, (A), false, (B), (short)0, (C), false, false)

#define DIM      1024
#define HEADS    16
#define DHEAD    64
#define INNER    2730
#define NTOK     1537          // 1 + 512 + 1 + 1023
#define NTOKP    1568          // padded to multiple of 32 (zero-filled pad)
#define NBATCH   2
#define BTOK     (NBATCH * NTOK)
#define CB_EOS   1025
#define CLEN     512
#define FLEN     1023

// A-fragment (16x32 f16): lane holds row lane%16; element i -> K = 2*koff+i,
// koff = (lane>>4)*8.
__device__ inline v16h afrag_h(const _Float16* __restrict__ p, int koff) {
  v8h lo = *(const v8h*)(p + 2 * koff);
  v8h hi = *(const v8h*)(p + 2 * koff + 8);
  v16h r;
#pragma unroll
  for (int i = 0; i < 8; ++i) { r[i] = lo[i]; r[i + 8] = hi[i]; }
  return r;
}
// B-fragment (32x16 f16): lane holds column lane%16; element i -> K = (lane>>4)*16+i
// => 16 contiguous K values starting at p (caller adds (lane>>4)*16).
__device__ inline v16h bfrag_h(const _Float16* __restrict__ p) {
  v8h lo = *(const v8h*)(p);
  v8h hi = *(const v8h*)(p + 8);
  v16h r;
#pragma unroll
  for (int i = 0; i < 8; ++i) { r[i] = lo[i]; r[i + 8] = hi[i]; }
  return r;
}

// ---------------------------------------------------------------------------
// Generic f32->f16 WMMA GEMM:  C[M,N] = A[M,K] * op(B) (+ R), fp32 accumulate.
// A row-major (row stride lda). B: K x N (ldb) if transB==0, else N x K (ldb).
// Register double-buffered staging; interior tiles use b64 loads / packed stores.
// All lda/ldb/offsets used here are even => float2 (8B) alignment holds.
// ---------------------------------------------------------------------------
#define BM 128
#define BN 128
#define BK 32

__global__ __launch_bounds__(256) void gemm_f16wmma(
    const float* __restrict__ A, long lda,
    const float* __restrict__ B, long ldb, int transB,
    float* __restrict__ C, long ldc,
    const float* __restrict__ R, long ldr,
    int M, int N, int K)
{
  __shared__ __align__(32) _Float16 As[BM][BK];
  __shared__ __align__(32) _Float16 Bs[BN][BK];   // stored [n][k]

  const int tid  = threadIdx.x;
  const int bm   = blockIdx.y, bn = blockIdx.x;
  const int wave = tid >> 5, lane = tid & 31;
  const int wm   = (wave & 3) * 32;   // 4 waves along M
  const int wn   = (wave >> 2) * 64;  // 2 waves along N
  const int rl = lane & 15, rhalf = lane >> 4, koff = rhalf * 8;

  const bool intM = (bm * BM + BM) <= M;
  const bool intN = (bn * BN + BN) <= N;

  float ra[16], rb[16];

  // ---- staging loads into registers (mapping shared with storeTiles) ----
  auto loadA = [&](int k0) {
    const bool fast = intM && ((k0 + BK) <= K);
    if (fast) {
#pragma unroll
      for (int i = 0; i < 8; ++i) {
        int u = tid + i * 256;              // 2048 float2 units
        int m = u >> 4, k2 = (u & 15) * 2;
        float2 v = *(const float2*)(A + (size_t)(bm * BM + m) * lda + (k0 + k2));
        ra[2 * i] = v.x; ra[2 * i + 1] = v.y;
      }
    } else {
#pragma unroll
      for (int i = 0; i < 8; ++i) {
        int u = tid + i * 256;
        int m = u >> 4, k2 = (u & 15) * 2;
        int gm = bm * BM + m;
#pragma unroll
        for (int j = 0; j < 2; ++j) {
          int gk = k0 + k2 + j;
          ra[2 * i + j] = (gm < M && gk < K) ? A[(size_t)gm * lda + gk] : 0.f;
        }
      }
    }
  };
  auto loadB = [&](int k0) {
    const bool fast = intN && ((k0 + BK) <= K);
    if (transB) {
      if (fast) {
#pragma unroll
        for (int i = 0; i < 8; ++i) {
          int u = tid + i * 256;
          int n = u >> 4, k2 = (u & 15) * 2;
          float2 v = *(const float2*)(B + (size_t)(bn * BN + n) * ldb + (k0 + k2));
          rb[2 * i] = v.x; rb[2 * i + 1] = v.y;
        }
      } else {
#pragma unroll
        for (int i = 0; i < 8; ++i) {
          int u = tid + i * 256;
          int n = u >> 4, k2 = (u & 15) * 2;
          int gn = bn * BN + n;
#pragma unroll
          for (int j = 0; j < 2; ++j) {
            int gk = k0 + k2 + j;
            rb[2 * i + j] = (gn < N && gk < K) ? B[(size_t)gn * ldb + gk] : 0.f;
          }
        }
      }
    } else {
      if (fast) {
#pragma unroll
        for (int i = 0; i < 8; ++i) {
          int u = tid + i * 256;
          int k = u >> 6, n2 = (u & 63) * 2;
          float2 v = *(const float2*)(B + (size_t)(k0 + k) * ldb + (bn * BN + n2));
          rb[2 * i] = v.x; rb[2 * i + 1] = v.y;
        }
      } else {
#pragma unroll
        for (int i = 0; i < 8; ++i) {
          int u = tid + i * 256;
          int k = u >> 6, n2 = (u & 63) * 2;
          int gk = k0 + k;
#pragma unroll
          for (int j = 0; j < 2; ++j) {
            int gn = bn * BN + n2 + j;
            rb[2 * i + j] = (gn < N && gk < K) ? B[(size_t)gk * ldb + gn] : 0.f;
          }
        }
      }
    }
  };
  auto storeTiles = [&]() {
#pragma unroll
    for (int i = 0; i < 8; ++i) {
      int u = tid + i * 256;
      int m = u >> 4, k2 = (u & 15) * 2;
      v2h t; t[0] = (_Float16)ra[2 * i]; t[1] = (_Float16)ra[2 * i + 1];
      *(v2h*)&As[m][k2] = t;
    }
    if (transB) {
#pragma unroll
      for (int i = 0; i < 8; ++i) {
        int u = tid + i * 256;
        int n = u >> 4, k2 = (u & 15) * 2;
        v2h t; t[0] = (_Float16)rb[2 * i]; t[1] = (_Float16)rb[2 * i + 1];
        *(v2h*)&Bs[n][k2] = t;
      }
    } else {
#pragma unroll
      for (int i = 0; i < 8; ++i) {
        int u = tid + i * 256;
        int k = u >> 6, n2 = (u & 63) * 2;
        Bs[n2][k]     = (_Float16)rb[2 * i];
        Bs[n2 + 1][k] = (_Float16)rb[2 * i + 1];
      }
    }
  };

  v8f acc[2][4] = {};

  loadA(0); loadB(0);
  for (int k0 = 0;;) {
    storeTiles();
    __syncthreads();
    const int k1 = k0 + BK;
    if (k1 < K) { loadA(k1); loadB(k1); }   // prefetch next tile into regs

    v16h af[2], bf[4];
#pragma unroll
    for (int mt = 0; mt < 2; ++mt)
      af[mt] = afrag_h(&As[wm + mt * 16 + rl][0], koff);
#pragma unroll
    for (int nt = 0; nt < 4; ++nt)
      bf[nt] = *(const v16h*)(&Bs[wn + nt * 16 + rl][rhalf * 16]);

#pragma unroll
    for (int mt = 0; mt < 2; ++mt)
#pragma unroll
      for (int nt = 0; nt < 4; ++nt)
        acc[mt][nt] = WMMA_F16(af[mt], bf[nt], acc[mt][nt]);

    __syncthreads();
    k0 = k1;
    if (k0 >= K) break;
  }

  // C layout: VGPR r -> M = r + (lane>>4)*8, N = lane&15
#pragma unroll
  for (int mt = 0; mt < 2; ++mt)
#pragma unroll
    for (int nt = 0; nt < 4; ++nt)
#pragma unroll
      for (int r = 0; r < 8; ++r) {
        int m = bm * BM + wm + mt * 16 + r + rhalf * 8;
        int n = bn * BN + wn + nt * 16 + rl;
        if (m < M && n < N) {
          float v = acc[mt][nt][r];
          if (R) v += R[(size_t)m * ldr + n];
          C[(size_t)m * ldc + n] = v;
        }
      }
}

// ---------------------------------------------------------------------------
// LayerNorm (population variance, eps 1e-5), one row per block.
// ---------------------------------------------------------------------------
__global__ __launch_bounds__(256) void ln_kernel(
    const float* __restrict__ in, const float* __restrict__ gamma,
    float* __restrict__ out, int D)
{
  const int row = blockIdx.x;
  const float* x = in + (size_t)row * D;
  float* y = out + (size_t)row * D;
  float s = 0.f, s2 = 0.f;
  for (int i = threadIdx.x; i < D; i += 256) { float v = x[i]; s += v; s2 += v * v; }
  __shared__ float r0[256], r1[256];
  r0[threadIdx.x] = s; r1[threadIdx.x] = s2;
  __syncthreads();
  for (int off = 128; off > 0; off >>= 1) {
    if (threadIdx.x < off) {
      r0[threadIdx.x] += r0[threadIdx.x + off];
      r1[threadIdx.x] += r1[threadIdx.x + off];
    }
    __syncthreads();
  }
  float mean = r0[0] / (float)D;
  float var  = r1[0] / (float)D - mean * mean;
  float rstd = rsqrtf(var + 1e-5f);
  for (int i = threadIdx.x; i < D; i += 256)
    y[i] = (x[i] - mean) * rstd * gamma[i];
}

// ---------------------------------------------------------------------------
// Token embedding gather: x[b, pos, :]
// ---------------------------------------------------------------------------
__global__ __launch_bounds__(256) void embed_kernel(
    const int* __restrict__ cids, const int* __restrict__ fids,
    const float* __restrict__ cemb, const float* __restrict__ cstart,
    const float* __restrict__ femb, const float* __restrict__ fstart,
    float* __restrict__ x)
{
  const int p = blockIdx.x;         // 0..BTOK-1
  const int b = p / NTOK, pos = p % NTOK;
  const float* src;
  if (pos == 0) {
    src = cstart;
  } else if (pos <= CLEN) {
    int t = pos - 1;
    int id = cids[b * CLEN + t] + (t & 3) * CB_EOS;
    src = cemb + (size_t)id * DIM;
  } else if (pos == CLEN + 1) {
    src = fstart;
  } else {
    int t = pos - (CLEN + 2);
    int id = fids[b * FLEN + t] + (t % 8) * CB_EOS;
    src = femb + (size_t)id * DIM;
  }
  float* dst = x + (size_t)p * DIM;
  for (int d = threadIdx.x; d < DIM; d += 256) dst[d] = src[d];
}

// ---------------------------------------------------------------------------
// Relative-position bias by distance: biasD[h*NTOK + dist] = table[bucket(dist)][h]
// ---------------------------------------------------------------------------
__global__ void bias_kernel(const float* __restrict__ table, float* __restrict__ biasD)
{
  const int dist = blockIdx.x;       // 0..NTOK-1
  const int h = threadIdx.x;         // 0..15
  int bk;
  if (dist < 16) {
    bk = dist;
  } else {
    int vl = 16 + (int)(logf((float)dist * (1.0f / 16.0f)) * (16.0f / logf(8.0f)));
    bk = vl < 31 ? vl : 31;
  }
  biasD[h * NTOK + dist] = table[bk * HEADS + h];
}

// ---------------------------------------------------------------------------
// Pack Q/K/V to f16 with zero-padded token dim (NTOKP):
//   qh [b][tp][1024]   (row-major f16)
//   kh [b][tp][64]     (token-major: B-fragment for QK^T is 16 contiguous d)
//   vT [b][d][tp]      (d-major:     B-fragment for P.V  is 16 contiguous tok)
// ---------------------------------------------------------------------------
__global__ __launch_bounds__(256) void pack_attn_kernel(
    const float* __restrict__ qbuf, const float* __restrict__ kvbuf,
    _Float16* __restrict__ qh, _Float16* __restrict__ kh, _Float16* __restrict__ vT)
{
  const int p = blockIdx.x;          // 0 .. NBATCH*NTOKP-1
  const int b = p / NTOKP, tp = p % NTOKP;
  const bool valid = tp < NTOK;
  const float* qr = qbuf + ((size_t)(b * NTOK + (valid ? tp : 0))) * DIM;
  _Float16* qo = qh + (size_t)p * DIM;
  for (int d = threadIdx.x; d < DIM; d += 256)
    qo[d] = valid ? (_Float16)qr[d] : (_Float16)0.f;
  if (threadIdx.x < DHEAD) {
    const int d = threadIdx.x;
    const float* kvr = kvbuf + ((size_t)(b * NTOK + (valid ? tp : 0))) * 128;
    kh[(size_t)p * DHEAD + d] = valid ? (_Float16)kvr[d] : (_Float16)0.f;
    vT[((size_t)(b * DHEAD + d)) * NTOKP + tp] =
        valid ? (_Float16)kvr[DHEAD + d] : (_Float16)0.f;
  }
}

// ---------------------------------------------------------------------------
// Flash-style causal attention, MQA (single 64-wide K/V head), WMMA QK^T + PV.
// One wave per (16-row q-tile, head, batch). All loads are unguarded 16B
// vector loads thanks to zero-padded f16 buffers.
// ---------------------------------------------------------------------------
__global__ __launch_bounds__(32) void attn_kernel(
    const _Float16* __restrict__ qh, const _Float16* __restrict__ kh,
    const _Float16* __restrict__ vT, const float* __restrict__ biasD,
    float* __restrict__ outbuf)
{
  const int itile = blockIdx.x, h = blockIdx.y, b = blockIdx.z;
  const int lane = threadIdx.x;
  const int rl = lane & 15, rhalf = lane >> 4, koff = rhalf * 8;
  const int i0 = itile * 16;

  __shared__ __align__(32) float    sS[16][32];
  __shared__ __align__(32) _Float16 sP[16][32];
  __shared__ float sScale[16];
  __shared__ float sInv[16];

  const _Float16* qp = qh + ((size_t)(b * NTOKP + i0 + rl)) * DIM + h * DHEAD;
  v16h aq0 = afrag_h(qp, koff);           // d = 0..31
  v16h aq1 = afrag_h(qp + 32, koff);      // d = 32..63

  v8f o[4] = {};
  float m_r = -__builtin_inff();
  float l_r = 0.f;

  for (int j0 = 0; j0 < i0 + 16; j0 += 32) {
    // ---- S = Q K^T over 32 keys, two 16x16 subtiles ----
#pragma unroll
    for (int sub = 0; sub < 2; ++sub) {
      const int tok = j0 + sub * 16 + rl;
      const _Float16* kp = kh + (size_t)(b * NTOKP + tok) * DHEAD;
      v16h bk0 = bfrag_h(kp + rhalf * 16);        // d = 0..31
      v16h bk1 = bfrag_h(kp + 32 + rhalf * 16);   // d = 32..63
      v8f s = {};
      s = WMMA_F16(aq0, bk0, s);
      s = WMMA_F16(aq1, bk1, s);
#pragma unroll
      for (int r = 0; r < 8; ++r) sS[r + rhalf * 8][sub * 16 + rl] = s[r];
    }
    __syncthreads();

    // ---- online softmax: lane r (<16) owns row i0+r ----
    if (lane < 16) {
      int irow = i0 + lane;
      float sv[32];
      float rowmax = -__builtin_inff();
#pragma unroll
      for (int j = 0; j < 32; ++j) {
        int jj = j0 + j;
        float v = -__builtin_inff();
        if (irow < NTOK && jj <= irow)
          v = sS[lane][j] * 0.125f + biasD[h * NTOK + (irow - jj)];
        sv[j] = v;
        rowmax = fmaxf(rowmax, v);
      }
      float mnew  = fmaxf(m_r, rowmax);
      float scale = (mnew == -__builtin_inff()) ? 1.f : expf(m_r - mnew);
      float ls = 0.f;
#pragma unroll
      for (int j = 0; j < 32; ++j) {
        float p = (sv[j] == -__builtin_inff()) ? 0.f : expf(sv[j] - mnew);
        ls += p;
        sP[lane][j] = (_Float16)p;
      }
      l_r = l_r * scale + ls;
      m_r = mnew;
      sScale[lane] = scale;
    }
    __syncthreads();

    // ---- rescale O, then O += P V ----
    float scv[8];
#pragma unroll
    for (int r = 0; r < 8; ++r) scv[r] = sScale[r + rhalf * 8];
#pragma unroll
    for (int dt = 0; dt < 4; ++dt)
#pragma unroll
      for (int r = 0; r < 8; ++r) o[dt][r] *= scv[r];

    v16h ap = afrag_h(&sP[rl][0], koff);   // P as A-fragment (16x32 over keys)
#pragma unroll
    for (int dt = 0; dt < 4; ++dt) {
      const _Float16* vp =
          vT + ((size_t)(b * DHEAD + dt * 16 + rl)) * NTOKP + j0 + rhalf * 16;
      v16h bv = bfrag_h(vp);
      o[dt] = WMMA_F16(ap, bv, o[dt]);
    }
    __syncthreads();
  }

  if (lane < 16) sInv[lane] = (l_r > 0.f) ? 1.f / l_r : 0.f;
  __syncthreads();
#pragma unroll
  for (int r = 0; r < 8; ++r) {
    int row = i0 + r + rhalf * 8;
    if (row < NTOK) {
      float inv = sInv[r + rhalf * 8];
      float* op = outbuf + ((size_t)(b * NTOK + row)) * DIM + h * DHEAD;
#pragma unroll
      for (int dt = 0; dt < 4; ++dt) op[dt * 16 + rl] = o[dt][r] * inv;
    }
  }
}

// ---------------------------------------------------------------------------
// FF middle: causal depthwise conv3 + split + exact GELU(gate)*a  -> z
// ---------------------------------------------------------------------------
__global__ __launch_bounds__(256) void ff_conv_kernel(
    const float* __restrict__ y, const float* __restrict__ w, float* __restrict__ z)
{
  const int t  = blockIdx.x;          // 0..BTOK-1
  const int tt = t % NTOK;            // position within batch
  const float* y0 = y + (size_t)t * (2 * INNER);
  const float* y1 = (tt >= 1) ? (y0 - (size_t)(2 * INNER)) : nullptr;
  const float* y2 = (tt >= 2) ? (y0 - (size_t)(4 * INNER)) : nullptr;
  for (int c = threadIdx.x; c < INNER; c += 256) {
    int ca = c, cg = c + INNER;
    float a = y0[ca] * w[ca * 3 + 2]
            + (y1 ? y1[ca] * w[ca * 3 + 1] : 0.f)
            + (y2 ? y2[ca] * w[ca * 3 + 0] : 0.f);
    float g = y0[cg] * w[cg * 3 + 2]
            + (y1 ? y1[cg] * w[cg * 3 + 1] : 0.f)
            + (y2 ? y2[cg] * w[cg * 3 + 0] : 0.f);
    float ge = 0.5f * g * (1.f + erff(g * 0.70710678118654752f));   // exact GELU
    z[(size_t)t * INNER + c] = ge * a;
  }
}

// ---------------------------------------------------------------------------
// Host side
// ---------------------------------------------------------------------------
static inline size_t align64(size_t n) { return (n + 63) & ~(size_t)63; }

static void launch_gemm(const float* A, long lda, const float* B, long ldb, int transB,
                        float* C, long ldc, const float* R, long ldr,
                        int M, int N, int K, hipStream_t s)
{
  dim3 g((N + BN - 1) / BN, (M + BM - 1) / BM);
  gemm_f16wmma<<<g, 256, 0, s>>>(A, lda, B, ldb, transB, C, ldc, R, ldr, M, N, K);
}

extern "C" void kernel_launch(void* const* d_in, const int* in_sizes, int n_in,
                              void* d_out, int out_size, void* d_ws, size_t ws_size,
                              hipStream_t stream)
{
  (void)in_sizes; (void)n_in; (void)out_size; (void)ws_size;

  // jax pytree (sorted-dict) flatten order:
  const int* cids = (const int*)d_in[0];
  const int* fids = (const int*)d_in[1];
  const float* coarse_emb   = (const float*)d_in[2];
  const float* coarse_start = (const float*)d_in[3];
  const float* coarse_w     = (const float*)d_in[4];
  const float* final_g      = (const float*)d_in[5];
  const float* fine_emb     = (const float*)d_in[6];
  const float* fine_start   = (const float*)d_in[7];
  const float* fine_w       = (const float*)d_in[8];
  // layers: per layer 9 leaves: norm_g, wkv, wo, wq, conv_w, ln1_g, ln2_g, w1, w2
  const float* rel_table    = (const float*)d_in[9 + 4 * 9];

  float* out = (float*)d_out;
  float* fout = out + (size_t)NBATCH * CLEN * CB_EOS;   // fine logits region

  // workspace layout (float units)
  float* ws = (float*)d_ws;
  size_t off = 0;
  const size_t SX  = align64((size_t)BTOK * DIM);
  const size_t SKV = align64((size_t)BTOK * 128);
  const size_t SB  = align64((size_t)HEADS * NTOK);
  const size_t SY  = align64((size_t)BTOK * 2 * INNER);
  const size_t SZ  = align64((size_t)BTOK * INNER);
  const size_t SQH = align64((size_t)NBATCH * NTOKP * DIM / 2);   // f16 in float units
  const size_t SKH = align64((size_t)NBATCH * NTOKP * DHEAD / 2);
  float* x     = ws + off; off += SX;
  float* xn    = ws + off; off += SX;
  float* qbuf  = ws + off; off += SX;
  float* aobuf = ws + off; off += SX;
  float* kvbuf = ws + off; off += SKV;
  float* biasD = ws + off; off += SB;
  float* ybuf  = ws + off; off += SY;
  float* zbuf  = ws + off; off += SZ;
  float* znbuf = ws + off; off += SZ;
  _Float16* qh = (_Float16*)(ws + off); off += SQH;
  _Float16* kh = (_Float16*)(ws + off); off += SKH;
  _Float16* vT = (_Float16*)(ws + off); off += SKH;

  // 1) embeddings + relative-position bias table
  embed_kernel<<<dim3(BTOK), 256, 0, stream>>>(cids, fids, coarse_emb, coarse_start,
                                               fine_emb, fine_start, x);
  bias_kernel<<<dim3(NTOK), HEADS, 0, stream>>>(rel_table, biasD);

  const int ITILES = (NTOK + 15) / 16;   // 97

  for (int l = 0; l < 4; ++l) {
    const int base = 9 + l * 9;
    const float* norm_g = (const float*)d_in[base + 0];
    const float* wkv    = (const float*)d_in[base + 1];
    const float* wo     = (const float*)d_in[base + 2];
    const float* wq     = (const float*)d_in[base + 3];
    const float* conv_w = (const float*)d_in[base + 4];
    const float* ln1_g  = (const float*)d_in[base + 5];
    const float* ln2_g  = (const float*)d_in[base + 6];
    const float* w1     = (const float*)d_in[base + 7];
    const float* w2     = (const float*)d_in[base + 8];

    // attention block
    ln_kernel<<<dim3(BTOK), 256, 0, stream>>>(x, norm_g, xn, DIM);
    launch_gemm(xn, DIM, wq,  DIM, 0, qbuf,  DIM, nullptr, 0, BTOK, DIM, DIM, stream);
    launch_gemm(xn, DIM, wkv, 128, 0, kvbuf, 128, nullptr, 0, BTOK, 128, DIM, stream);
    pack_attn_kernel<<<dim3(NBATCH * NTOKP), 256, 0, stream>>>(qbuf, kvbuf, qh, kh, vT);
    attn_kernel<<<dim3(ITILES, HEADS, NBATCH), 32, 0, stream>>>(qh, kh, vT, biasD, aobuf);
    launch_gemm(aobuf, DIM, wo, DIM, 0, x, DIM, x, DIM, BTOK, DIM, DIM, stream); // +res

    // feed-forward block
    ln_kernel<<<dim3(BTOK), 256, 0, stream>>>(x, ln1_g, xn, DIM);
    launch_gemm(xn, DIM, w1, 2 * INNER, 0, ybuf, 2 * INNER, nullptr, 0,
                BTOK, 2 * INNER, DIM, stream);
    ff_conv_kernel<<<dim3(BTOK), 256, 0, stream>>>(ybuf, conv_w, zbuf);
    ln_kernel<<<dim3(BTOK), 256, 0, stream>>>(zbuf, ln2_g, znbuf, INNER);
    launch_gemm(znbuf, INNER, w2, DIM, 0, x, DIM, x, DIM, BTOK, DIM, INNER, stream); // +res
  }

  // final LN
  ln_kernel<<<dim3(BTOK), 256, 0, stream>>>(x, final_g, xn, DIM);

  // coarse logits: positions 0..511, quantizer q = pos % 4; W[q] is (1025,1024) -> B^T
  for (int b = 0; b < NBATCH; ++b)
    for (int q = 0; q < 4; ++q)
      launch_gemm(xn + ((size_t)b * NTOK + q) * DIM, 4 * DIM,
                  coarse_w + (size_t)q * CB_EOS * DIM, DIM, 1,
                  out + ((size_t)b * CLEN + q) * CB_EOS, 4 * CB_EOS,
                  nullptr, 0, CLEN / 4, CB_EOS, DIM, stream);

  // fine logits: positions 513..1536 (1024 rows), quantizer q = idx % 8
  for (int b = 0; b < NBATCH; ++b)
    for (int q = 0; q < 8; ++q)
      launch_gemm(xn + ((size_t)b * NTOK + (CLEN + 1) + q) * DIM, 8 * DIM,
                  fine_w + (size_t)q * CB_EOS * DIM, DIM, 1,
                  fout + ((size_t)b * 1024 + q) * CB_EOS, 8 * CB_EOS,
                  nullptr, 0, 1024 / 8, CB_EOS, DIM, stream);
}
